// PatchSA_40338332844433
// MI455X (gfx1250) — matrix operand order repaired and checked by reading, not compiled, not run on hardware
//
#include <hip/hip_runtime.h>

typedef __attribute__((ext_vector_type(16))) _Float16 v16h;
typedef __attribute__((ext_vector_type(8)))  float    v8f;

#define DIM   256
#define HEADS 8
#define HD    32     // head dim
#define PW    8      // window size
#define ST    4      // stride
#define NW    24     // windows per side
#define IMG   96
#define NPOS  64     // PW*PW positions per window
#define HW    (IMG*IMG)   // 9216 pixels

__device__ __forceinline__ int imin(int a, int b) { return a < b ? a : b; }

// ---------------------------------------------------------------------------
// Kernel 0: expand pos_table (225 x 8) into bias8[h][row][col] (8 x 64 x 64,
// 128 KB). Runs once; removes all bias index math from the attention kernel.
// ---------------------------------------------------------------------------
__global__ void bias_expand_kernel(const float* __restrict__ pos_table,
                                   float* __restrict__ bias8)
{
    const int i = blockIdx.x * blockDim.x + threadIdx.x;
    if (i >= HEADS * NPOS * NPOS) return;
    const int h  = i / (NPOS * NPOS);
    const int rc = i % (NPOS * NPOS);
    const int row = rc >> 6, col = rc & 63;
    const int qi = row >> 3, qj = row & 7;
    const int ki = col >> 3, kj = col & 7;
    const int bidx = (qi - ki + 7) * 15 + (qj - kj + 7);
    bias8[i] = pos_table[bidx * HEADS + h];
}

// ---------------------------------------------------------------------------
// Kernel 1: per-(window, head) attention. One wave32 per block (EXEC all-1s
// for WMMA). Q/K staged [pos][ch], V staged transposed [ch][pos] so every
// WMMA operand build is a contiguous 16-half run (2x ds_load_b128).
// PRE_BIAS=true reads the expanded bias table with linear addressing.
// ---------------------------------------------------------------------------
template <bool PRE_BIAS>
__global__ __launch_bounds__(32)
void win_attn_kernel(const float* __restrict__ x,
                     const float* __restrict__ qkv_w,
                     const float* __restrict__ qkv_b,
                     const float* __restrict__ pos_table,
                     const float* __restrict__ bias8,
                     float* __restrict__ attn_out)
{
    __shared__ _Float16 sQ [NPOS * HD];   // [pos][ch]   4 KB
    __shared__ _Float16 sK [NPOS * HD];   // [pos][ch]   4 KB
    __shared__ _Float16 sVt[HD * NPOS];   // [ch][pos]   4 KB  (transposed)
    __shared__ float    sS [NPOS * NPOS]; // scores     16 KB
    __shared__ _Float16 sP [NPOS * NPOS]; // probs       8 KB

    const int lane = threadIdx.x;
    int wid = blockIdx.x;
    const int h  = wid % HEADS; wid /= HEADS;
    const int wx = wid % NW;    wid /= NW;
    const int wy = wid % NW;
    const int b  = wid / NW;

    const float rsc = 0.17677669529663687f; // 1/sqrt(32)
    const float* __restrict__ bh = PRE_BIAS ? (bias8 + h * NPOS * NPOS) : nullptr;

    // ---- stage QKV (diagonal qkv: per-channel scale+bias), fp32 -> f16 ----
    for (int i = lane; i < NPOS * HD; i += 32) {
        const int c  = i >> 6;          // 0..31 channel within head
        const int p  = i & 63;          // window position
        const int ry = p >> 3, rx = p & 7;
        const int gy = imin(wy * ST + ry, IMG - 1);   // edge padding
        const int gx = imin(wx * ST + rx, IMG - 1);
        const int ch = h * HD + c;
        const float xv = x[((b * DIM + ch) * IMG + gy) * IMG + gx];
        const float qv = xv * qkv_w[ch]           + qkv_b[ch];
        const float kv = xv * qkv_w[DIM + ch]     + qkv_b[DIM + ch];
        const float vv = xv * qkv_w[2 * DIM + ch] + qkv_b[2 * DIM + ch];
        sQ [p * HD + c]   = (_Float16)(qv * rsc);
        sK [p * HD + c]   = (_Float16)kv;
        sVt[c * NPOS + p] = (_Float16)vv;      // transposed store
    }
    __syncthreads();

    const int mlane  = lane & 15;
    const int hiHalf = (lane < 16) ? 0 : 8;    // C-layout M-base
    const int hiK16  = (lane < 16) ? 0 : 16;   // B-layout K-base

    // ---- S = Q * K^T + bias : 4x4 tiles of 16x16, K=32 in one WMMA --------
    #pragma unroll
    for (int mt = 0; mt < 4; ++mt) {
        // A tile: Q rows mt*16.., contiguous k = 16h + e, same order as B
        v16h a;
        const int arow = mt * 16 + mlane;
        #pragma unroll
        for (int e = 0; e < 8; ++e) {
            a[e]     = sQ[arow * HD + hiK16 + e];
            a[e + 8] = sQ[arow * HD + hiK16 + 8 + e];
        }
        #pragma unroll
        for (int nt = 0; nt < 4; ++nt) {
            // B tile: B[c][n] = K[n][c], contiguous k = 16h + e, same order as A
            v16h bm;
            const int bcol = nt * 16 + mlane;
            #pragma unroll
            for (int e = 0; e < 16; ++e)
                bm[e] = sK[bcol * HD + hiK16 + e];

            v8f c8 = {};
            c8 = __builtin_amdgcn_wmma_f32_16x16x32_f16(
                     false, a, false, bm, (short)0, c8, false, false);

            // spill C tile to LDS, fusing relative-position bias
            #pragma unroll
            for (int r = 0; r < 8; ++r) {
                const int row = mt * 16 + r + hiHalf;
                const int col = nt * 16 + mlane;
                float bv;
                if (PRE_BIAS) {
                    bv = bh[row * NPOS + col];      // linear, clause-friendly
                } else {
                    const int qi = row >> 3, qj = row & 7;
                    const int ki = col >> 3, kj = col & 7;
                    const int bidx = (qi - ki + 7) * 15 + (qj - kj + 7);
                    bv = pos_table[bidx * HEADS + h];
                }
                sS[row * NPOS + col] = c8[r] + bv;
            }
        }
    }
    __syncthreads();

    // ---- row softmax: 2 rows per lane ------------------------------------
    for (int rr = lane; rr < NPOS; rr += 32) {
        float mx = -1e30f;
        for (int j = 0; j < NPOS; ++j) mx = fmaxf(mx, sS[rr * NPOS + j]);
        float sum = 0.f;
        for (int j = 0; j < NPOS; ++j) {
            float e = __expf(sS[rr * NPOS + j] - mx);
            sS[rr * NPOS + j] = e;
            sum += e;
        }
        const float inv = 1.f / sum;
        for (int j = 0; j < NPOS; ++j)
            sP[rr * NPOS + j] = (_Float16)(sS[rr * NPOS + j] * inv);
    }
    __syncthreads();

    // ---- O = P(64x64) * V(64x32): accumulate over two K=32 steps ---------
    #pragma unroll
    for (int mt = 0; mt < 4; ++mt) {
        #pragma unroll
        for (int nt = 0; nt < 2; ++nt) {
            v8f acc = {};
            #pragma unroll
            for (int ks = 0; ks < 2; ++ks) {
                v16h a;
                const int arow = mt * 16 + mlane;
                const int kb = ks * 32 + hiK16;
                #pragma unroll
                for (int e = 0; e < 8; ++e) {
                    a[e]     = sP[arow * NPOS + kb + e];
                    a[e + 8] = sP[arow * NPOS + kb + 8 + e];
                }
                // B tile: B[kk][n] = V[kk][n] = sVt[n][kk] -> contiguous in kk
                v16h bm;
                const int kk   = ks * 32 + hiK16;
                const int ncol = nt * 16 + mlane;
                #pragma unroll
                for (int e = 0; e < 16; ++e)
                    bm[e] = sVt[ncol * NPOS + kk + e];

                acc = __builtin_amdgcn_wmma_f32_16x16x32_f16(
                          false, a, false, bm, (short)0, acc, false, false);
            }
            // keep only the top-left 4x4 of the window (unique owner per pixel)
            #pragma unroll
            for (int r = 0; r < 8; ++r) {
                const int row = mt * 16 + r + hiHalf;
                const int ry = row >> 3, rx = row & 7;
                if (ry < ST && rx < ST) {
                    const int gy = wy * ST + ry;
                    const int gx = wx * ST + rx;
                    const int ch = h * HD + nt * 16 + mlane;
                    attn_out[((b * DIM + ch) * IMG + gy) * IMG + gx] = acc[r];
                }
            }
        }
    }
}

// ---------------------------------------------------------------------------
// Kernel 2: channel projection out[b,o,yx] = sum_c out_w[o,c] * attn[b,c,yx].
// One wave32 per 16x16 (o, pixel) tile, 8 accumulating WMMAs over K=256.
// attn intermediate (19 MB) is L2-resident (192 MB L2). B loads are
// lane-coalesced (consecutive pixels across lanes for each fixed channel).
// ---------------------------------------------------------------------------
__global__ __launch_bounds__(32)
void proj_kernel(const float* __restrict__ out_w,
                 const float* __restrict__ attn,
                 float* __restrict__ out)
{
    const int lane = threadIdx.x;
    int t = blockIdx.x;
    const int nt = t % (HW / 16);  t /= (HW / 16);   // pixel tile 0..575
    const int mt = t % (DIM / 16);                   // out-channel tile 0..15
    const int b  = t / (DIM / 16);

    const int mlane  = lane & 15;
    const int hiHalf = (lane < 16) ? 0 : 8;
    const int hiK16  = (lane < 16) ? 0 : 16;

    v8f acc = {};
    #pragma unroll
    for (int ks = 0; ks < DIM / 32; ++ks) {
        v16h a;
        const int m  = mt * 16 + mlane;
        const int kb = ks * 32 + hiK16;
        #pragma unroll
        for (int e = 0; e < 8; ++e) {
            a[e]     = (_Float16)out_w[m * DIM + kb + e];
            a[e + 8] = (_Float16)out_w[m * DIM + kb + 8 + e];
        }
        v16h bm;
        const int kk = ks * 32 + hiK16;
        const int n  = nt * 16 + mlane;
        #pragma unroll
        for (int e = 0; e < 16; ++e)
            bm[e] = (_Float16)attn[(b * DIM + kk + e) * HW + n];

        acc = __builtin_amdgcn_wmma_f32_16x16x32_f16(
                  false, a, false, bm, (short)0, acc, false, false);
    }
    #pragma unroll
    for (int r = 0; r < 8; ++r) {
        const int row = mt * 16 + r + hiHalf;
        out[(b * DIM + row) * HW + nt * 16 + mlane] = acc[r];
    }
}

extern "C" void kernel_launch(void* const* d_in, const int* in_sizes, int n_in,
                              void* d_out, int out_size, void* d_ws, size_t ws_size,
                              hipStream_t stream) {
    (void)in_sizes; (void)n_in; (void)out_size;
    const float* x         = (const float*)d_in[0];
    const float* qkv_w     = (const float*)d_in[1];
    const float* qkv_b     = (const float*)d_in[2];
    const float* out_w     = (const float*)d_in[3];
    const float* pos_table = (const float*)d_in[4];
    float* out  = (float*)d_out;

    const size_t attn_bytes = (size_t)2 * DIM * HW * sizeof(float);   // 18.9 MB
    const size_t bias_bytes = (size_t)HEADS * NPOS * NPOS * sizeof(float); // 128 KB
    float* attn  = (float*)d_ws;
    float* bias8 = (float*)((char*)d_ws + attn_bytes);
    const bool use_bias_table = (ws_size >= attn_bytes + bias_bytes);

    dim3 blk(32);
    if (use_bias_table) {
        bias_expand_kernel<<<(HEADS * NPOS * NPOS + 255) / 256, 256, 0, stream>>>(
            pos_table, bias8);
        win_attn_kernel<true><<<2 * NW * NW * HEADS, blk, 0, stream>>>(
            x, qkv_w, qkv_b, pos_table, bias8, attn);
    } else {
        win_attn_kernel<false><<<2 * NW * NW * HEADS, blk, 0, stream>>>(
            x, qkv_w, qkv_b, pos_table, nullptr, attn);
    }
    proj_kernel<<<2 * (DIM / 16) * (HW / 16), blk, 0, stream>>>(
        out_w, attn, out);
}
